// Attention_38792144617606
// MI455X (gfx1250) — matrix operand order repaired and checked by reading, not compiled, not run on hardware
//
#include <hip/hip_runtime.h>
#include <hip/hip_bf16.h>
#include <stdint.h>

// ---------------------------------------------------------------------------
// CDNA5 (gfx1250) wave32 WMMA types
// ---------------------------------------------------------------------------
typedef __attribute__((ext_vector_type(16))) __bf16 v16bf;
typedef __attribute__((ext_vector_type(8)))  float  v8f;

union Frag16 {                // 16 bf16 halves == 2 x uint4 (32 bytes)
    v16bf v;
    uint4 u4[2];
    uint16_t h[16];
};

// Native f32->bf16 (backend selects v_cvt_(pk_)bf16_f32 on gfx1250)
__device__ __forceinline__ uint16_t f2bfbits(float f) {
    union { __bf16 b; uint16_t u; } t;
    t.b = (__bf16)f;
    return t.u;
}

// Problem constants
#define Bsz 8
#define Hn  8
#define NQ  1024
#define NK  1024
#define Dh  32      // per-head dim

// Padded LDS strides (halves) - conflict-free for 16-lane b128 fragment reads
#define KSTR  40     // K rows: 80B  -> bank stride 20 (gcd 4 with 64)
#define VSTR  1032   // Vt rows: 2064B -> bank stride 4
#define PSTR  40     // P rows: 80B

// ---------------------------------------------------------------------------
// One-shot converters (run once, tiny cost)
// ---------------------------------------------------------------------------
__global__ void cvt_bf16_kernel(const float* __restrict__ src,
                                uint16_t* __restrict__ dst)
{
    const size_t i = ((size_t)blockIdx.x * blockDim.x + threadIdx.x) * 8;
    float4 a = *(const float4*)(src + i);
    float4 b = *(const float4*)(src + i + 4);
    union { uint16_t h[8]; uint4 u; } o;
    o.h[0] = f2bfbits(a.x); o.h[1] = f2bfbits(a.y);
    o.h[2] = f2bfbits(a.z); o.h[3] = f2bfbits(a.w);
    o.h[4] = f2bfbits(b.x); o.h[5] = f2bfbits(b.y);
    o.h[6] = f2bfbits(b.z); o.h[7] = f2bfbits(b.w);
    *(uint4*)(dst + i) = o.u;
}

// W f32 row-major [k=256][n=256] -> Wt bf16 col-major [n=256][k=256]
__global__ void cvtT_bf16_kernel(const float* __restrict__ W,
                                 uint16_t* __restrict__ Wt)
{
    const int n = blockIdx.x;     // 256
    const int k = threadIdx.x;    // 256
    Wt[n * 256 + k] = f2bfbits(W[(size_t)k * 256 + n]);
}

// ---------------------------------------------------------------------------
// GEMM: C[m,n] = sum_k A[m,k] * Wt[n,k]   (M=8192, N=256, K=256), bf16 in.
// block = 128 threads (4 waves); each wave computes a 16x64 strip (4 tiles).
// ---------------------------------------------------------------------------
__global__ void gemm256_wmma(const uint16_t* __restrict__ A,
                             const uint16_t* __restrict__ Wt,
                             const float* __restrict__ bvec,
                             float* __restrict__ outF,
                             uint16_t* __restrict__ outH,
                             int mode, float scale)
{
    const int wave = threadIdx.x >> 5;
    const int lane = threadIdx.x & 31;
    const int half = lane >> 4;
    const int l16  = lane & 15;

    const int m0    = blockIdx.x * 16;
    const int nbase = wave * 64;

    v8f acc[4] = {{}, {}, {}, {}};

    for (int kk = 0; kk < 256; kk += 32) {
        Frag16 af;
        {
            const uint16_t* ap = A + (size_t)(m0 + l16) * 256 + kk + half * 8;
            af.u4[0] = *(const uint4*)(ap);
            af.u4[1] = *(const uint4*)(ap + 16);
        }
#pragma unroll
        for (int j = 0; j < 4; ++j) {
            Frag16 bf;
            const uint16_t* wp = Wt + (size_t)(nbase + j * 16 + l16) * 256 + kk + half * 8;
            bf.u4[0] = *(const uint4*)(wp);
            bf.u4[1] = *(const uint4*)(wp + 16);
            acc[j] = __builtin_amdgcn_wmma_f32_16x16x32_bf16(
                         false, af.v, false, bf.v, (short)0, acc[j], false, false);
        }
    }

#pragma unroll
    for (int j = 0; j < 4; ++j) {
        const int n = nbase + j * 16 + l16;
#pragma unroll
        for (int r = 0; r < 8; ++r) {
            const int m = m0 + r + half * 8;
            float v = acc[j][r] * scale;
            if (mode <= 2) {
                const int b = m >> 10, row = m & 1023;
                const int h = n >> 5,  c   = n & 31;
                outH[((size_t)((b * Hn + h) * NQ + row) << 5) + c] = f2bfbits(v);
            } else if (mode == 3) {
                v = v + bvec[n];
                v = 1.0f / (1.0f + __expf(-v));
                outF[(size_t)m * 256 + n] = v;
            } else {
                outF[(size_t)m * 256 + n] = v + bvec[n];
            }
        }
    }
}

// ---------------------------------------------------------------------------
// Flash attention per (b,h). Whole head's K + V^T staged in LDS (CDNA5: up to
// 320KB/workgroup). K staged via async global->LDS DMA (ASYNCcnt).
// block = 256 threads (8 waves), each wave owns 16 queries; grid (NQ/128, B*H)
// Dynamic LDS: 80KB sK (padded) + 64.5KB sVt (padded) + 10KB sP = 154.5KB.
// ---------------------------------------------------------------------------
__global__ void attn_kernel(const uint16_t* __restrict__ Qh,   // [B,H,NQ,32] bf16
                            const uint16_t* __restrict__ Kh,   // [B,H,NK,32] bf16
                            const uint16_t* __restrict__ Vh,   // [B,H,NK,32] bf16
                            const float*    __restrict__ bias, // [B,1,NQ,NK]
                            const float*    __restrict__ nbias,// [H,NQ,NK]
                            const float*    __restrict__ gate, // [B,NQ,256] f32
                            uint16_t*       __restrict__ wab)  // [B,NQ,256] bf16
{
    extern __shared__ __align__(16) char smem[];
    uint16_t* sK  = (uint16_t*)(smem);             // [key][c] rows of KSTR halves
    uint16_t* sVt = (uint16_t*)(smem + 81920);     // [c][key] rows of VSTR halves
    uint16_t* sP  = (uint16_t*)(smem + 147968);    // per-wave 16 rows of PSTR

    const int wave = threadIdx.x >> 5;
    const int lane = threadIdx.x & 31;
    const int half = lane >> 4;
    const int l16  = lane & 15;

    const int bh = blockIdx.y;
    const int b  = bh >> 3;
    const int h  = bh & 7;
    const int q0 = blockIdx.x * 128 + wave * 16;

    const uint16_t* Kbase = Kh + ((size_t)bh * NK) * Dh;
    const uint16_t* Vbase = Vh + ((size_t)bh * NK) * Dh;
    const uint16_t* Qbase = Qh + ((size_t)bh * NQ) * Dh;
    const float*    biasQ = bias  + (size_t)b * NQ * NK;
    const float*    nbQ   = nbias + (size_t)h * NQ * NK;

    // ---- stage K via async global->LDS DMA: 4 x b128 per padded 80B row
    {
        const uint32_t ldsK = (uint32_t)(uintptr_t)sK;   // low 32 bits = LDS offset
#pragma unroll
        for (int i = 0; i < 16; ++i) {
            const uint32_t op  = i * 256 + threadIdx.x;  // 4096 ops total
            const uint32_t row = op >> 2;
            const uint32_t seg = op & 3;
            const uint64_t ga  = (uint64_t)(uintptr_t)(Kbase + row * Dh + seg * 8);
            const uint32_t la  = ldsK + row * (KSTR * 2) + seg * 16;
            asm volatile("global_load_async_to_lds_b128 %0, %1, off"
                         :: "v"(la), "v"(ga) : "memory");
        }
    }
    // ---- stage V transposed [c][key] through VGPRs
    for (int i = 0; i < 16; ++i) {
        const uint32_t op  = i * 256 + threadIdx.x;
        const uint32_t key = op >> 2;
        const uint32_t c   = (op & 3) * 8;
        union { uint4 u; uint16_t s[8]; } vv;
        vv.u = *(const uint4*)(Vbase + key * Dh + c);
#pragma unroll
        for (int jj = 0; jj < 8; ++jj)
            sVt[(c + jj) * VSTR + key] = vv.s[jj];
    }
    asm volatile("s_wait_asynccnt 0" ::: "memory");
    __syncthreads();

    // ---- Q fragment (A operand, 16x32)
    Frag16 qa;
    {
        const uint16_t* qp = Qbase + (size_t)(q0 + l16) * Dh + half * 8;
        qa.u4[0] = *(const uint4*)(qp);
        qa.u4[1] = *(const uint4*)(qp + 16);
    }

    float mrow[8], lsum[8];
#pragma unroll
    for (int r = 0; r < 8; ++r) { mrow[r] = -1e30f; lsum[r] = 0.0f; }
    v8f acc0 = {}, acc1 = {};
    uint16_t* sPw = sP + wave * 16 * PSTR;

    for (int kc = 0; kc < NK; kc += 32) {
        // ---- logits: two 16-key tiles, GEMM-K = 32 (head dim)
        v8f s0, s1;
        {
            const v8f z = {};
            Frag16 kb0, kb1;
            const uint16_t* kp0 = &sK[(size_t)(kc + 0  + l16) * KSTR + half * 8];
            const uint16_t* kp1 = &sK[(size_t)(kc + 16 + l16) * KSTR + half * 8];
            kb0.u4[0] = *(const uint4*)(kp0); kb0.u4[1] = *(const uint4*)(kp0 + 16);
            kb1.u4[0] = *(const uint4*)(kp1); kb1.u4[1] = *(const uint4*)(kp1 + 16);
            s0 = __builtin_amdgcn_wmma_f32_16x16x32_bf16(
                     false, qa.v, false, kb0.v, (short)0, z, false, false);
            s1 = __builtin_amdgcn_wmma_f32_16x16x32_bf16(
                     false, qa.v, false, kb1.v, (short)0, z, false, false);
        }

        // ---- add biases: streaming single-use f32 -> non-temporal loads
#pragma unroll
        for (int r = 0; r < 8; ++r) {
            const int q = q0 + r + half * 8;
            const size_t rowoff = (size_t)q * NK + kc;
            s0[r] += __builtin_nontemporal_load(biasQ + rowoff + 0  + l16)
                   + __builtin_nontemporal_load(nbQ   + rowoff + 0  + l16);
            s1[r] += __builtin_nontemporal_load(biasQ + rowoff + 16 + l16)
                   + __builtin_nontemporal_load(nbQ   + rowoff + 16 + l16);
        }

        // ---- online softmax (row stats across 16 lanes of each half)
        float tmax[8], rsum[8];
#pragma unroll
        for (int r = 0; r < 8; ++r) tmax[r] = fmaxf(s0[r], s1[r]);
#pragma unroll
        for (int off = 1; off < 16; off <<= 1)
#pragma unroll
            for (int r = 0; r < 8; ++r)
                tmax[r] = fmaxf(tmax[r], __shfl_xor(tmax[r], off, 32));

#pragma unroll
        for (int r = 0; r < 8; ++r) {
            const float mnew = fmaxf(mrow[r], tmax[r]);
            const float sc   = __expf(mrow[r] - mnew);
            mrow[r] = mnew;
            const float p0 = __expf(s0[r] - mnew);
            const float p1 = __expf(s1[r] - mnew);
            s0[r] = p0; s1[r] = p1;
            rsum[r] = p0 + p1;
            acc0[r] *= sc; acc1[r] *= sc;
            lsum[r] *= sc;
        }
#pragma unroll
        for (int off = 1; off < 16; off <<= 1)
#pragma unroll
            for (int r = 0; r < 8; ++r)
                rsum[r] += __shfl_xor(rsum[r], off, 32);
#pragma unroll
        for (int r = 0; r < 8; ++r) lsum[r] += rsum[r];

        // ---- transpose P (D layout) -> A layout via per-wave LDS scratch
#pragma unroll
        for (int r = 0; r < 8; ++r) {
            const int m = r + half * 8;
            sPw[m * PSTR + 0  + l16] = f2bfbits(s0[r]);
            sPw[m * PSTR + 16 + l16] = f2bfbits(s1[r]);
        }
        asm volatile("s_wait_dscnt 0" ::: "memory");   // wave-local DS RAW

        Frag16 pa;
        {
            const uint16_t* pp = sPw + l16 * PSTR + half * 8;
            pa.u4[0] = *(const uint4*)(pp);
            pa.u4[1] = *(const uint4*)(pp + 16);
        }

        // ---- PV accumulate: B operand from transposed V
        {
            Frag16 vb0, vb1;
            const uint16_t* vp0 = &sVt[(size_t)(0  + l16) * VSTR + kc + half * 8];
            const uint16_t* vp1 = &sVt[(size_t)(16 + l16) * VSTR + kc + half * 8];
            vb0.u4[0] = *(const uint4*)(vp0); vb0.u4[1] = *(const uint4*)(vp0 + 16);
            vb1.u4[0] = *(const uint4*)(vp1); vb1.u4[1] = *(const uint4*)(vp1 + 16);
            acc0 = __builtin_amdgcn_wmma_f32_16x16x32_bf16(
                       false, pa.v, false, vb0.v, (short)0, acc0, false, false);
            acc1 = __builtin_amdgcn_wmma_f32_16x16x32_bf16(
                       false, pa.v, false, vb1.v, (short)0, acc1, false, false);
        }
    }

    // ---- epilogue: normalize, gate, store weighted_avg as bf16
#pragma unroll
    for (int r = 0; r < 8; ++r) {
        const float inv = 1.0f / lsum[r];
        const int q = q0 + r + half * 8;
        const size_t base = ((size_t)(b * NQ + q)) * 256 + h * Dh;
        wab[base + 0  + l16] = f2bfbits(acc0[r] * inv * gate[base + 0  + l16]);
        wab[base + 16 + l16] = f2bfbits(acc1[r] * inv * gate[base + 16 + l16]);
    }
}

// ---------------------------------------------------------------------------
// Host-side launch
// ---------------------------------------------------------------------------
extern "C" void kernel_launch(void* const* d_in, const int* in_sizes, int n_in,
                              void* d_out, int out_size, void* d_ws, size_t ws_size,
                              hipStream_t stream)
{
    const float* q_data   = (const float*)d_in[0];
    const float* m_data   = (const float*)d_in[1];
    const float* bias     = (const float*)d_in[2];
    const float* nbias    = (const float*)d_in[3];
    const float* query_w  = (const float*)d_in[4];
    const float* key_w    = (const float*)d_in[5];
    const float* value_w  = (const float*)d_in[6];
    const float* gating_w = (const float*)d_in[7];
    const float* gating_b = (const float*)d_in[8];
    const float* output_w = (const float*)d_in[9];
    const float* output_b = (const float*)d_in[10];
    float* out = (float*)d_out;

    char* ws = (char*)d_ws;
    const size_t MB = 1ull << 20;
    uint16_t* qb    = (uint16_t*)(ws);                 // 4 MiB bf16 q_data
    uint16_t* mb    = (uint16_t*)(ws + 4 * MB);        // 4 MiB bf16 m_data
    uint16_t* wt_q  = (uint16_t*)(ws + 8 * MB);        // 128 KiB each, col-major
    uint16_t* wt_k  = (uint16_t*)(ws + 8 * MB + 1 * 131072);
    uint16_t* wt_v  = (uint16_t*)(ws + 8 * MB + 2 * 131072);
    uint16_t* wt_g  = (uint16_t*)(ws + 8 * MB + 3 * 131072);
    uint16_t* wt_o  = (uint16_t*)(ws + 8 * MB + 4 * 131072);
    uint16_t* Qh    = (uint16_t*)(ws + 9 * MB);        // 4 MiB [B,H,NQ,32]
    uint16_t* Kh    = (uint16_t*)(ws + 13 * MB);       // 4 MiB
    uint16_t* Vh    = (uint16_t*)(ws + 17 * MB);       // 4 MiB
    float*    gate  = (float*)   (ws + 21 * MB);       // 8 MiB f32 [B,NQ,256]
    uint16_t* wab   = (uint16_t*)(ws + 29 * MB);       // 4 MiB bf16 [B,NQ,256]

    // one-shot bf16 conversions
    cvt_bf16_kernel<<<dim3(1024), 256, 0, stream>>>(q_data, qb);   // 2M elems
    cvt_bf16_kernel<<<dim3(1024), 256, 0, stream>>>(m_data, mb);
    cvtT_bf16_kernel<<<dim3(256), 256, 0, stream>>>(query_w,  wt_q);
    cvtT_bf16_kernel<<<dim3(256), 256, 0, stream>>>(key_w,    wt_k);
    cvtT_bf16_kernel<<<dim3(256), 256, 0, stream>>>(value_w,  wt_v);
    cvtT_bf16_kernel<<<dim3(256), 256, 0, stream>>>(gating_w, wt_g);
    cvtT_bf16_kernel<<<dim3(256), 256, 0, stream>>>(output_w, wt_o);

    const dim3 gblk(128);
    const dim3 ggrid(Bsz * NQ / 16);                   // 512 blocks, wave = 16x64
    const float qscale = 0.17677669529663689f;         // 32^-0.5

    gemm256_wmma<<<ggrid, gblk, 0, stream>>>(qb, wt_q, nullptr,  nullptr, Qh, 0, qscale);
    gemm256_wmma<<<ggrid, gblk, 0, stream>>>(mb, wt_k, nullptr,  nullptr, Kh, 1, 1.0f);
    gemm256_wmma<<<ggrid, gblk, 0, stream>>>(mb, wt_v, nullptr,  nullptr, Vh, 2, 1.0f);
    gemm256_wmma<<<ggrid, gblk, 0, stream>>>(qb, wt_g, gating_b, gate, nullptr, 3, 1.0f);

    // dynamic LDS: 81920 (sK) + 66048 (sVt) + 10240 (sP) = 158208 bytes
    attn_kernel<<<dim3(NQ / 128, Bsz * Hn), 256, 158208, stream>>>(
        Qh, Kh, Vh, bias, nbias, gate, wab);

    gemm256_wmma<<<ggrid, gblk, 0, stream>>>(wab, wt_o, output_b, out, nullptr, 4, 1.0f);
}
